// MyUAGCRN_78400333021796
// MI455X (gfx1250) — matrix operand order repaired and checked by reading, not compiled, not run on hardware
//
#include <hip/hip_runtime.h>
#include <math.h>

// ---------------------------------------------------------------------------
// Problem constants (from reference): B=64, P=Q=12, N=325, D=64
// Padded: M -> 336 (21x16 tiles), K(node dim) -> 352 (11x32 tiles)
// ---------------------------------------------------------------------------
#define BB   64
#define PP   12
#define QQ   12
#define NN   325
#define DD   64
#define NPM  336   // node dim padded for WMMA M (21*16)
#define NPK  352   // node dim padded for WMMA K (11*32)
#define TT   24    // P+Q

typedef __bf16 bf16;
typedef __attribute__((ext_vector_type(16))) __bf16 v16bf;
typedef __attribute__((ext_vector_type(8)))  __bf16 v8bf;
typedef __attribute__((ext_vector_type(8)))  float  v8f;

__device__ __forceinline__ bf16 f2bf(float f) {
    unsigned int u = __builtin_bit_cast(unsigned int, f);
    u += 0x7FFFu + ((u >> 16) & 1u);               // round-to-nearest-even
    return __builtin_bit_cast(bf16, (unsigned short)(u >> 16));
}
__device__ __forceinline__ float sigmoidf_(float x) { return 1.0f / (1.0f + expf(-x)); }

// ---------------------------------------------------------------------------
// WMMA fragment helpers (A/B use this code's contiguous k order, not the ISA table order; wave32)
// A 16x32 bf16: lane L (half=L>>4, m=L&15): e -> K=half*16+e  (16 contiguous K)
// B 32x16 bf16: lane L (n=L&15, half=L>>4): e -> K=half*16+e  (16 contiguous K!)
// C/D 16x16 f32: VGPR r -> M=r+8*half, N=lane&15
//
// B operands are stored pre-swizzled: each 32(K)x16(N) tile = 32 lanes x 16
// contiguous bf16  ->  fragment load = one 32B contiguous load per lane.
// ---------------------------------------------------------------------------
__device__ __forceinline__ v16bf loadFragA(const bf16* __restrict__ base, int ldk) {
    int lane = threadIdx.x & 31;
    int half = lane >> 4, m = lane & 15;
    const bf16* p = base + m * ldk + half * 16;
    v8bf lo = *(const v8bf*)(p);
    v8bf hi = *(const v8bf*)(p + 8);
    v16bf a;
#pragma unroll
    for (int e = 0; e < 8; ++e) { a[e] = lo[e]; a[e + 8] = hi[e]; }
    return a;
}
__device__ __forceinline__ v16bf loadFragBswz(const bf16* __restrict__ tile) {
    int lane = threadIdx.x & 31;
    return *(const v16bf*)(tile + lane * 16);
}
__device__ __forceinline__ v8f wmma_bf16(v16bf a, v16bf b, v8f c) {
    return __builtin_amdgcn_wmma_f32_16x16x32_bf16(false, a, false, b, (short)0, c, false, false);
}
// swizzled offset of element (k=n_node, n=f) inside one (K=NPK x F) B matrix
__device__ __forceinline__ size_t swzoff(int n, int f, int nftiles) {
    int kt = n >> 5, ft = f >> 4;
    int lane = (((n >> 4) & 1) << 4) | (f & 15);
    int e = n & 15;
    return ((size_t)(kt * nftiles + ft)) * 512 + lane * 16 + e;
}

// ---------------------------------------------------------------------------
// GEMM 1: out[b] = S (336x352 bf16 padded) @ v[b] (352x128, B-swizzled) -> bf16
// grid (21, 4, B), block 32: each wave does TWO 16x16 tiles sharing A
// ---------------------------------------------------------------------------
__global__ __launch_bounds__(32) void spmm_bf16(const bf16* __restrict__ S,
                                                const bf16* __restrict__ Vswz,
                                                bf16* __restrict__ O) {
    int mt = blockIdx.x, fp = blockIdx.y, b = blockIdx.z;
    const bf16* Vb = Vswz + (size_t)b * NPK * 128;
    bf16*       Ob = O    + (size_t)b * NPM * 128;
    v8f acc0 = {}, acc1 = {};
    for (int kt = 0; kt < NPK / 32; ++kt) {
        if (kt + 1 < NPK / 32)
            __builtin_prefetch(Vb + ((size_t)((kt + 1) * 8 + 2 * fp)) * 512, 0, 1);
        v16bf A  = loadFragA(S + (size_t)mt * 16 * NPK + kt * 32, NPK);
        v16bf B0 = loadFragBswz(Vb + ((size_t)(kt * 8 + 2 * fp + 0)) * 512);
        v16bf B1 = loadFragBswz(Vb + ((size_t)(kt * 8 + 2 * fp + 1)) * 512);
        acc0 = wmma_bf16(A, B0, acc0);
        acc1 = wmma_bf16(A, B1, acc1);
    }
    int lane = threadIdx.x & 31, half = lane >> 4, n = lane & 15;
    bf16* op = Ob + (size_t)mt * 16 * 128 + (2 * fp) * 16 + n;
#pragma unroll
    for (int r = 0; r < 8; ++r) {
        op[(size_t)(r + 8 * half) * 128]      = f2bf(acc0[r]);
        op[(size_t)(r + 8 * half) * 128 + 16] = f2bf(acc1[r]);
    }
}

// ---------------------------------------------------------------------------
// GEMM 2: out = [v | S0v | S1v] (Mx384, 3 chunk ptrs, row-major A)
//               @ W (384 x ldw, B-swizzled) -> f32
// grid (21, ldw/32, B), block 32: two 16x16 tiles per wave
// ---------------------------------------------------------------------------
__global__ __launch_bounds__(32) void xsw_bf16(const bf16* __restrict__ A0,
                                               const bf16* __restrict__ A1,
                                               const bf16* __restrict__ A2,
                                               const bf16* __restrict__ Wswz,
                                               float* __restrict__ O, int ldw) {
    int mt = blockIdx.x, fp = blockIdx.y, b = blockIdx.z;
    int nft = ldw >> 4;
    const bf16* chunks[3] = { A0 + (size_t)b * NPK * 128,
                              A1 + (size_t)b * NPM * 128,
                              A2 + (size_t)b * NPM * 128 };
    v8f acc0 = {}, acc1 = {};
    for (int kt = 0; kt < 12; ++kt) {
        const bf16* Ab = chunks[kt >> 2];
        v16bf A  = loadFragA(Ab + (size_t)mt * 16 * 128 + (kt & 3) * 32, 128);
        v16bf B0 = loadFragBswz(Wswz + ((size_t)(kt * nft + 2 * fp + 0)) * 512);
        v16bf B1 = loadFragBswz(Wswz + ((size_t)(kt * nft + 2 * fp + 1)) * 512);
        acc0 = wmma_bf16(A, B0, acc0);
        acc1 = wmma_bf16(A, B1, acc1);
    }
    float* Ob = O + (size_t)b * NPM * ldw + (size_t)mt * 16 * ldw + (2 * fp) * 16;
    int lane = threadIdx.x & 31, half = lane >> 4, n = lane & 15;
#pragma unroll
    for (int r = 0; r < 8; ++r) {
        Ob[(size_t)(r + 8 * half) * ldw + n]      = acc0[r];
        Ob[(size_t)(r + 8 * half) * ldw + n + 16] = acc1[r];
    }
}

// ---------------------------------------------------------------------------
// Prep kernels
// ---------------------------------------------------------------------------
__global__ void pad_S_kernel(const float* __restrict__ S, bf16* __restrict__ dst) {
    int i = blockIdx.x * blockDim.x + threadIdx.x;
    if (i >= NPM * NPK) return;
    int m = i / NPK, k = i % NPK;
    float v = (m < NN && k < NN) ? S[m * NN + k] : 0.0f;
    dst[i] = f2bf(v);
}
// W (Kx ldw f32, K=384) -> B-fragment-swizzled bf16
__global__ void swz_w_kernel(const float* __restrict__ W, bf16* __restrict__ dst, int ldw) {
    int total = 12 * (ldw >> 4) * 512;   // (K/32) * (ldw/16) * 512
    int i = blockIdx.x * blockDim.x + threadIdx.x;
    if (i >= total) return;
    int tile = i >> 9, r = i & 511;
    int lane = r >> 4, e = r & 15;
    int nft = ldw >> 4;
    int kt = tile / nft, ft = tile % nft;
    int k = kt * 32 + (lane >> 4) * 16 + e;
    int n = ft * 16 + (lane & 15);
    dst[i] = f2bf(W[k * ldw + n]);
}
__global__ void zero_f32_kernel(float* __restrict__ p, int n) {
    int i = blockIdx.x * blockDim.x + threadIdx.x;
    if (i < n) p[i] = 0.0f;
}

// te[b,t,:] = relu(W_te1[wd] + W_te1[7+td] + b_te1) @ W_te2 + b_te2   (one block/row)
__global__ __launch_bounds__(64) void te_kernel(const int* __restrict__ TE,
                                                const float* __restrict__ W1,
                                                const float* __restrict__ b1,
                                                const float* __restrict__ W2,
                                                const float* __restrict__ b2,
                                                float* __restrict__ te) {
    __shared__ float hsm[64];
    int row = blockIdx.x;         // b*24 + t
    int j = threadIdx.x;
    int wd = TE[row * 2 + 0], td = TE[row * 2 + 1];
    float h = W1[wd * 64 + j] + W1[(7 + td) * 64 + j] + b1[j];
    hsm[j] = fmaxf(h, 0.0f);
    __syncthreads();
    float acc = b2[j];
    for (int k = 0; k < 64; ++k) acc += hsm[k] * W2[k * 64 + j];
    te[(size_t)row * 64 + j] = acc;
}

// xenc[t,b,n,:] = bf16( (relu(X*Win1+bin1)@Win2+bin2) + te[b,t] + E_se[n] )
__global__ __launch_bounds__(64) void xenc_kernel(const float* __restrict__ X,
                                                  const float* __restrict__ te,
                                                  const float* __restrict__ Ese,
                                                  const float* __restrict__ Win1,
                                                  const float* __restrict__ bin1,
                                                  const float* __restrict__ Win2,
                                                  const float* __restrict__ bin2,
                                                  bf16* __restrict__ xenc) {
    __shared__ float asm_[64];
    long idx = blockIdx.x;        // t*B*N + b*N + n
    int n = (int)(idx % NN);
    long r = idx / NN;
    int b = (int)(r % BB);
    int t = (int)(r / BB);
    int j = threadIdx.x;
    float x = X[((size_t)b * PP + t) * NN + n];
    asm_[j] = fmaxf(x * Win1[j] + bin1[j], 0.0f);
    __syncthreads();
    float acc = bin2[j];
    for (int k = 0; k < 64; ++k) acc += asm_[k] * Win2[k * 64 + j];
    acc += te[((size_t)b * TT + t) * 64 + j] + Ese[(size_t)n * 64 + j];
    xenc[(size_t)idx * 64 + j] = f2bf(acc);
}

// xdec[t,b,n,:] = bf16( te[b,P+t] + E_se[n] )
__global__ void xdec_kernel(const float* __restrict__ te, const float* __restrict__ Ese,
                            bf16* __restrict__ xdec) {
    long i = (long)blockIdx.x * blockDim.x + threadIdx.x;
    long total = (long)QQ * BB * NN * 64;
    if (i >= total) return;
    int d = (int)(i % 64);
    int n = (int)((i / 64) % NN);
    int b = (int)((i / (64L * NN)) % BB);
    int t = (int)(i / (64L * NN * BB));
    float v = te[((size_t)b * TT + PP + t) * 64 + d] + Ese[(size_t)n * 64 + d];
    xdec[i] = f2bf(v);
}

// vbuf init: [x0 | 0], zero pad rows; writes row-major copy AND B-swizzled copy
__global__ void init_v_kernel(const bf16* __restrict__ x0, bf16* __restrict__ vbuf,
                              bf16* __restrict__ vswz) {
    long i = (long)blockIdx.x * blockDim.x + threadIdx.x;
    long total = (long)BB * NPK * 128;
    if (i >= total) return;
    int f = (int)(i % 128);
    int n = (int)((i / 128) % NPK);
    int b = (int)(i / (128L * NPK));
    bf16 v = f2bf(0.0f);
    if (n < NN && f < 64) v = x0[((size_t)b * NN + n) * 64 + f];
    vbuf[i] = v;
    vswz[(size_t)b * NPK * 128 + swzoff(n, f, 8)] = v;
}

// ---------------------------------------------------------------------------
// GRU gate kernels (write both row-major and swizzled v copies)
// ---------------------------------------------------------------------------
__global__ void gateA_kernel(const float* __restrict__ ru, const float* __restrict__ bg,
                             const float* __restrict__ h, bf16* __restrict__ vbuf,
                             bf16* __restrict__ vswz) {
    long i = (long)blockIdx.x * blockDim.x + threadIdx.x;
    long total = (long)BB * NN * 64;
    if (i >= total) return;
    int f = (int)(i % 64);
    int n = (int)((i / 64) % NN);
    int b = (int)(i / (64L * NN));
    float r  = sigmoidf_(ru[((size_t)b * NPM + n) * 128 + f] + bg[f]);
    float hv = h[((size_t)b * NPM + n) * 64 + f];
    bf16 v = f2bf(r * hv);
    vbuf[((size_t)b * NPK + n) * 128 + 64 + f] = v;
    vswz[(size_t)b * NPK * 128 + swzoff(n, 64 + f, 8)] = v;
}

__global__ void gateB_kernel(const float* __restrict__ cpre, const float* __restrict__ bc,
                             const float* __restrict__ ru, const float* __restrict__ bg,
                             float* __restrict__ h, bf16* __restrict__ vbuf,
                             bf16* __restrict__ vswz, const bf16* __restrict__ xnext) {
    long i = (long)blockIdx.x * blockDim.x + threadIdx.x;
    long total = (long)BB * NN * 64;
    if (i >= total) return;
    int f = (int)(i % 64);
    int n = (int)((i / 64) % NN);
    int b = (int)(i / (64L * NN));
    float c = tanhf(cpre[((size_t)b * NPM + n) * 64 + f] + bc[f]);
    float u = sigmoidf_(ru[((size_t)b * NPM + n) * 128 + 64 + f] + bg[64 + f]);
    size_t hi = ((size_t)b * NPM + n) * 64 + f;
    float hn = u * h[hi] + (1.0f - u) * c;
    h[hi] = hn;
    bf16 hb = f2bf(hn);
    size_t vi = ((size_t)b * NPK + n) * 128;
    size_t si = (size_t)b * NPK * 128;
    vbuf[vi + 64 + f] = hb;
    vswz[si + swzoff(n, 64 + f, 8)] = hb;
    if (xnext) {
        bf16 xb = xnext[((size_t)b * NN + n) * 64 + f];
        vbuf[vi + f] = xb;
        vswz[si + swzoff(n, f, 8)] = xb;
    }
}

// out[b,t,n] = relu(h@W1+b1)@W2 + b2   (one wave per (b,n) row, wave32 reduce)
__global__ __launch_bounds__(256) void proj_kernel(const float* __restrict__ h,
                                                   const float* __restrict__ W1,
                                                   const float* __restrict__ b1,
                                                   const float* __restrict__ W2,
                                                   const float* __restrict__ b2,
                                                   float* __restrict__ out, int t) {
    int wid = threadIdx.x >> 5, lane = threadIdx.x & 31;
    int row = blockIdx.x * 8 + wid;
    if (row >= BB * NN) return;
    int b = row / NN, n = row % NN;
    const float* hp = h + ((size_t)b * NPM + n) * 64;
    int j = lane;
    float a0 = 0.0f, a1 = 0.0f;
    for (int k = 0; k < 64; ++k) {
        float hv = hp[k];
        a0 += hv * W1[k * 64 + j];
        a1 += hv * W1[k * 64 + j + 32];
    }
    float o = fmaxf(a0 + b1[j], 0.0f) * W2[j] + fmaxf(a1 + b1[j + 32], 0.0f) * W2[j + 32];
#pragma unroll
    for (int off = 16; off > 0; off >>= 1) o += __shfl_xor(o, off, 32);
    if (lane == 0) out[((size_t)b * QQ + t) * NN + n] = o + b2[0];
}

// ---------------------------------------------------------------------------
// Workspace layout (bytes; everything 256B-aligned)
// ---------------------------------------------------------------------------
#define OFF_S0    ((size_t)0)
#define OFF_S1    (OFF_S0 + (size_t)NPM * NPK * 2)
#define OFF_WGE   (OFF_S1 + (size_t)NPM * NPK * 2)
#define OFF_WCE   (OFF_WGE + (size_t)384 * 128 * 2)
#define OFF_WGD   (OFF_WCE + (size_t)384 * 64 * 2)
#define OFF_WCD   (OFF_WGD + (size_t)384 * 128 * 2)
#define OFF_TE    (OFF_WCD + (size_t)384 * 64 * 2)
#define OFF_XENC  (OFF_TE + (size_t)BB * TT * 64 * 4)
#define OFF_XDEC  (OFF_XENC + (size_t)PP * BB * NN * 64 * 2)
#define OFF_H     (OFF_XDEC + (size_t)QQ * BB * NN * 64 * 2)
#define OFF_V     (OFF_H + (size_t)BB * NPM * 64 * 4)
#define OFF_VSWZ  (OFF_V + (size_t)BB * NPK * 128 * 2)
#define OFF_S0V   (OFF_VSWZ + (size_t)BB * NPK * 128 * 2)
#define OFF_S1V   (OFF_S0V + (size_t)BB * NPM * 128 * 2)
#define OFF_RU    (OFF_S1V + (size_t)BB * NPM * 128 * 2)
#define OFF_C     (OFF_RU + (size_t)BB * NPM * 128 * 4)

extern "C" void kernel_launch(void* const* d_in, const int* in_sizes, int n_in,
                              void* d_out, int out_size, void* d_ws, size_t ws_size,
                              hipStream_t stream) {
    const float* X      = (const float*)d_in[0];
    const int*   TE     = (const int*)d_in[1];
    const float* S0     = (const float*)d_in[2];
    const float* S1     = (const float*)d_in[3];
    const float* W_te1  = (const float*)d_in[4];
    const float* b_te1  = (const float*)d_in[5];
    const float* W_te2  = (const float*)d_in[6];
    const float* b_te2  = (const float*)d_in[7];
    const float* E_se   = (const float*)d_in[8];
    const float* W_in1  = (const float*)d_in[9];
    const float* b_in1  = (const float*)d_in[10];
    const float* W_in2  = (const float*)d_in[11];
    const float* b_in2  = (const float*)d_in[12];
    const float* enc_Wg = (const float*)d_in[13];
    const float* enc_bg = (const float*)d_in[14];
    const float* enc_Wc = (const float*)d_in[15];
    const float* enc_bc = (const float*)d_in[16];
    const float* dec_Wg = (const float*)d_in[17];
    const float* dec_bg = (const float*)d_in[18];
    const float* dec_Wc = (const float*)d_in[19];
    const float* dec_bc = (const float*)d_in[20];
    const float* W_out1 = (const float*)d_in[21];
    const float* b_out1 = (const float*)d_in[22];
    const float* W_out2 = (const float*)d_in[23];
    const float* b_out2 = (const float*)d_in[24];
    float* out = (float*)d_out;

    char* ws = (char*)d_ws;
    bf16*  Sbf0 = (bf16*)(ws + OFF_S0);
    bf16*  Sbf1 = (bf16*)(ws + OFF_S1);
    bf16*  WgE  = (bf16*)(ws + OFF_WGE);
    bf16*  WcE  = (bf16*)(ws + OFF_WCE);
    bf16*  WgD  = (bf16*)(ws + OFF_WGD);
    bf16*  WcD  = (bf16*)(ws + OFF_WCD);
    float* te   = (float*)(ws + OFF_TE);
    bf16*  xenc = (bf16*)(ws + OFF_XENC);
    bf16*  xdec = (bf16*)(ws + OFF_XDEC);
    float* hbuf = (float*)(ws + OFF_H);
    bf16*  vbuf = (bf16*)(ws + OFF_V);
    bf16*  vswz = (bf16*)(ws + OFF_VSWZ);
    bf16*  s0v  = (bf16*)(ws + OFF_S0V);
    bf16*  s1v  = (bf16*)(ws + OFF_S1V);
    float* ru   = (float*)(ws + OFF_RU);
    float* cpre = (float*)(ws + OFF_C);

    // -------- Prep --------
    {
        int n = NPM * NPK;
        pad_S_kernel<<<(n + 255) / 256, 256, 0, stream>>>(S0, Sbf0);
        pad_S_kernel<<<(n + 255) / 256, 256, 0, stream>>>(S1, Sbf1);
    }
    swz_w_kernel<<<(384 * 128 + 255) / 256, 256, 0, stream>>>(enc_Wg, WgE, 128);
    swz_w_kernel<<<(384 * 64  + 255) / 256, 256, 0, stream>>>(enc_Wc, WcE, 64);
    swz_w_kernel<<<(384 * 128 + 255) / 256, 256, 0, stream>>>(dec_Wg, WgD, 128);
    swz_w_kernel<<<(384 * 64  + 255) / 256, 256, 0, stream>>>(dec_Wc, WcD, 64);
    te_kernel<<<BB * TT, 64, 0, stream>>>(TE, W_te1, b_te1, W_te2, b_te2, te);
    xenc_kernel<<<PP * BB * NN, 64, 0, stream>>>(X, te, E_se, W_in1, b_in1, W_in2, b_in2, xenc);
    {
        long n = (long)QQ * BB * NN * 64;
        xdec_kernel<<<(unsigned)((n + 255) / 256), 256, 0, stream>>>(te, E_se, xdec);
    }
    zero_f32_kernel<<<(BB * NPM * 64 + 255) / 256, 256, 0, stream>>>(hbuf, BB * NPM * 64);
    {
        long n = (long)BB * NPK * 128;
        init_v_kernel<<<(unsigned)((n + 255) / 256), 256, 0, stream>>>(xenc, vbuf, vswz);
    }

    // -------- DCGRU steps --------
    const dim3 gS(NPM / 16, 128 / 32, BB);   // spmm + gate-gemm grid (2 tiles/wave)
    const dim3 gC(NPM / 16, 64 / 32, BB);    // candidate-gemm grid
    const int gateBlocks = (BB * NN * 64 + 255) / 256;

    auto step = [&](const bf16* Wg, const float* bg, const bf16* Wc, const float* bc,
                    const bf16* xnext, int proj_t) {
        // phase 1: gate
        spmm_bf16<<<gS, 32, 0, stream>>>(Sbf0, vswz, s0v);
        spmm_bf16<<<gS, 32, 0, stream>>>(Sbf1, vswz, s1v);
        xsw_bf16<<<gS, 32, 0, stream>>>(vbuf, s0v, s1v, Wg, ru, 128);
        gateA_kernel<<<gateBlocks, 256, 0, stream>>>(ru, bg, hbuf, vbuf, vswz);
        // phase 2: candidate (v = [x | r*h])
        spmm_bf16<<<gS, 32, 0, stream>>>(Sbf0, vswz, s0v);
        spmm_bf16<<<gS, 32, 0, stream>>>(Sbf1, vswz, s1v);
        xsw_bf16<<<gC, 32, 0, stream>>>(vbuf, s0v, s1v, Wc, cpre, 64);
        gateB_kernel<<<gateBlocks, 256, 0, stream>>>(cpre, bc, ru, bg, hbuf, vbuf, vswz, xnext);
        if (proj_t >= 0)
            proj_kernel<<<(BB * NN + 7) / 8, 256, 0, stream>>>(hbuf, W_out1, b_out1,
                                                               W_out2, b_out2, out, proj_t);
    };

    const size_t xstride = (size_t)BB * NN * 64;
    for (int t = 0; t < PP; ++t) {
        const bf16* xnext = (t < PP - 1) ? (xenc + (size_t)(t + 1) * xstride) : xdec;
        step(WgE, enc_bg, WcE, enc_bc, xnext, -1);
    }
    for (int t = 0; t < QQ; ++t) {
        const bf16* xnext = (t < QQ - 1) ? (xdec + (size_t)(t + 1) * xstride) : nullptr;
        step(WgD, dec_bg, WcD, dec_bc, xnext, t);
    }
}
